// PoincareAttentionLayer_56788057588129
// MI455X (gfx1250) — matrix operand order repaired and checked by reading, not compiled, not run on hardware
//
#include <hip/hip_runtime.h>
#include <hip/hip_bf16.h>
#include <math.h>

#define EPSF 1e-6f
#define BALL_EPSF 1e-5f

typedef __attribute__((ext_vector_type(16))) __bf16 v16bf;
typedef __attribute__((ext_vector_type(8)))  __bf16 v8bf;
typedef __attribute__((ext_vector_type(8)))  float  v8f;

union BU { uint4 u; v8bf b; };

__device__ __forceinline__ float atanh_f(float u) {
    return 0.5f * __logf((1.f + u) / (1.f - u));
}

// ---- block reductions (blockDim.x == 256) ----
__device__ __forceinline__ float blk_sum(float v, float* sm) {
    int t = threadIdx.x;
    sm[t] = v; __syncthreads();
    for (int o = 128; o > 0; o >>= 1) { if (t < o) sm[t] += sm[t + o]; __syncthreads(); }
    float r = sm[0]; __syncthreads();
    return r;
}
__device__ __forceinline__ float blk_max(float v, float* sm) {
    int t = threadIdx.x;
    sm[t] = v; __syncthreads();
    for (int o = 128; o > 0; o >>= 1) { if (t < o) sm[t] = fmaxf(sm[t], sm[t + o]); __syncthreads(); }
    float r = sm[0]; __syncthreads();
    return r;
}

// ---- fp32 -> bf16 conversion ----
__global__ void k_cvt_bf16(const float* __restrict__ in, __bf16* __restrict__ out, int n) {
    int i = blockIdx.x * blockDim.x + threadIdx.x;
    int stride = gridDim.x * blockDim.x;
    for (; i < n; i += stride) out[i] = (__bf16)in[i];
}

// ---- enc_k / enc_v preprocessing: k2, bf16(k), lambda, bf16(v*lambda), lambda-1 ----
__global__ __launch_bounds__(256)
void k_prep_kv(const float* __restrict__ k, const float* __restrict__ v,
               __bf16* __restrict__ kbf, __bf16* __restrict__ vlbf,
               float* __restrict__ k2o, float* __restrict__ lamm1) {
    __shared__ float sm[256];
    const int row = blockIdx.x;                 // b*S + s, E = 1024
    const float* kr = k + (size_t)row * 1024;
    const float* vr = v + (size_t)row * 1024;
    float kv[4], vv[4];
    float ks = 0.f, vs = 0.f;
#pragma unroll
    for (int j = 0; j < 4; j++) {
        int idx = threadIdx.x + j * 256;
        kv[j] = kr[idx]; vv[j] = vr[idx];
        ks += kv[j] * kv[j]; vs += vv[j] * vv[j];
    }
    float k2v = blk_sum(ks, sm);
    float v2v = blk_sum(vs, sm);
    float lam = 2.f / fmaxf(1.f - v2v, EPSF);
#pragma unroll
    for (int j = 0; j < 4; j++) {
        int idx = threadIdx.x + j * 256;
        kbf [(size_t)row * 1024 + idx] = (__bf16)kv[j];
        vlbf[(size_t)row * 1024 + idx] = (__bf16)(vv[j] * lam);
    }
    if (threadIdx.x == 0) { k2o[row] = k2v; lamm1[row] = lam - 1.f; }
}

// ---- bf16 tile transpose, per batch [1024,1024] -> [1024,1024]^T ----
__global__ __launch_bounds__(256)
void k_transpose_bf(const __bf16* __restrict__ in, __bf16* __restrict__ out) {
    __shared__ __bf16 tile[64][68];
    const int b = blockIdx.z;
    const int s0 = blockIdx.y * 64, e0 = blockIdx.x * 64;
    const __bf16* src = in + (size_t)b * 1024 * 1024;
    __bf16* dst = out + (size_t)b * 1024 * 1024;
    const int tx = threadIdx.x & 15;       // 16 x 4-element groups
    const int ty = threadIdx.x >> 4;       // 16 rows per pass
#pragma unroll
    for (int r = 0; r < 64; r += 16) {
        const __bf16* p = src + (size_t)(s0 + r + ty) * 1024 + e0 + tx * 4;
#pragma unroll
        for (int j = 0; j < 4; j++) tile[r + ty][tx * 4 + j] = p[j];
    }
    __syncthreads();
#pragma unroll
    for (int r = 0; r < 64; r += 16) {
        __bf16* p = dst + (size_t)(e0 + r + ty) * 1024 + s0 + tx * 4;
#pragma unroll
        for (int j = 0; j < 4; j++) p[j] = tile[tx * 4 + j][r + ty];
    }
}

// ---- WMMA fragment helpers (bf16 16x16x32, wave32 layouts per CDNA5 ISA) ----
__device__ __forceinline__ v16bf load_a_frag(const __bf16* arow, int akoff) {
    BU a0, a1;
    a0.u = *(const uint4*)(arow + akoff);        // K = akoff .. akoff+7
    a1.u = *(const uint4*)(arow + 16 + akoff);   // K = 16+akoff ..
    v16bf af;
#pragma unroll
    for (int j = 0; j < 8; j++) { af[j] = a0.b[j]; af[j + 8] = a1.b[j]; }
    return af;
}
__device__ __forceinline__ v16bf load_b_frag_nt(const __bf16* brow, int bkoff) {
    BU b0, b1;
    b0.u = *(const uint4*)(brow + bkoff);
    b1.u = *(const uint4*)(brow + bkoff + 16);
    v16bf bf;
#pragma unroll
    for (int j = 0; j < 8; j++) { bf[j] = b0.b[j]; bf[j + 8] = b1.b[j]; }
    return bf;
}
#define WMMA_BF16(a, b, c) \
    __builtin_amdgcn_wmma_f32_16x16x32_bf16(false, (a), false, (b), (short)0, (c), false, false)

// ---- NT GEMM, 32x32 wave tile, 64x128 block tile, optional batching ----
// C[m,n] = sum_k A[m,k] * Bm[n,k]
__global__ __launch_bounds__(256)
void k_gemm_nt(const __bf16* __restrict__ A, const __bf16* __restrict__ Bm,
               float* __restrict__ C, int N, int K,
               size_t sA, size_t sB, size_t sC) {
    const size_t zb = blockIdx.z;
    A += zb * sA; Bm += zb * sB; C += zb * sC;
    const int lane = threadIdx.x & 31;
    const int wave = threadIdx.x >> 5;
    const int wm = wave & 1, wn = wave >> 1;       // 2x4 waves -> 64(M) x 128(N)
    const int m0 = blockIdx.y * 64 + wm * 32;
    const int n0 = blockIdx.x * 128 + wn * 32;
    const int l15 = lane & 15, half = lane >> 4;
    const int akoff = half * 8, bkoff = half * 8;
    const __bf16* ar0 = A + (size_t)(m0 + l15) * K;
    const __bf16* ar1 = ar0 + (size_t)16 * K;
    const __bf16* br0 = Bm + (size_t)(n0 + l15) * K;
    const __bf16* br1 = br0 + (size_t)16 * K;
    v8f acc00 = {}, acc01 = {}, acc10 = {}, acc11 = {};
    for (int k0 = 0; k0 < K; k0 += 32) {
        v16bf a0 = load_a_frag(ar0 + k0, akoff);
        v16bf a1 = load_a_frag(ar1 + k0, akoff);
        v16bf b0 = load_b_frag_nt(br0 + k0, bkoff);
        v16bf b1 = load_b_frag_nt(br1 + k0, bkoff);
        acc00 = WMMA_BF16(a0, b0, acc00);
        acc01 = WMMA_BF16(a0, b1, acc01);
        acc10 = WMMA_BF16(a1, b0, acc10);
        acc11 = WMMA_BF16(a1, b1, acc11);
    }
    float* c0 = C + (size_t)(m0 + half * 8) * N + n0 + l15;
#pragma unroll
    for (int i = 0; i < 8; i++) {
        c0[(size_t)i * N]            = acc00[i];
        c0[(size_t)i * N + 16]       = acc01[i];
        c0[(size_t)(i + 16) * N]     = acc10[i];
        c0[(size_t)(i + 16) * N + 16] = acc11[i];
    }
}

// ---- GEMM2: batched NT h @ enc_k^T with fused Poincare-distance epilogue ----
__global__ __launch_bounds__(256)
void k_gemm_scores(const __bf16* __restrict__ H, const __bf16* __restrict__ Kb,
                   const float* __restrict__ q2a, const float* __restrict__ k2a,
                   const float* __restrict__ scale, float* __restrict__ Sc) {
    const int b = blockIdx.z;
    const int T = 1024, S = 1024, E = 1024;
    const __bf16* A  = H  + (size_t)b * T * E;
    const __bf16* Bm = Kb + (size_t)b * S * E;
    float* C = Sc + (size_t)b * T * S;
    const float* q2 = q2a + (size_t)b * T;
    const float* k2 = k2a + (size_t)b * S;

    const int lane = threadIdx.x & 31;
    const int wave = threadIdx.x >> 5;
    const int wm = wave & 1, wn = wave >> 1;
    const int m0 = blockIdx.y * 64 + wm * 32;
    const int n0 = blockIdx.x * 128 + wn * 32;
    const int l15 = lane & 15, half = lane >> 4;
    const int akoff = half * 8, bkoff = half * 8;
    const __bf16* ar0 = A + (size_t)(m0 + l15) * E;
    const __bf16* ar1 = ar0 + (size_t)16 * E;
    const __bf16* br0 = Bm + (size_t)(n0 + l15) * E;
    const __bf16* br1 = br0 + (size_t)16 * E;
    v8f acc00 = {}, acc01 = {}, acc10 = {}, acc11 = {};
    for (int k0 = 0; k0 < E; k0 += 32) {
        v16bf a0 = load_a_frag(ar0 + k0, akoff);
        v16bf a1 = load_a_frag(ar1 + k0, akoff);
        v16bf b0 = load_b_frag_nt(br0 + k0, bkoff);
        v16bf b1 = load_b_frag_nt(br1 + k0, bkoff);
        acc00 = WMMA_BF16(a0, b0, acc00);
        acc01 = WMMA_BF16(a0, b1, acc01);
        acc10 = WMMA_BF16(a1, b0, acc10);
        acc11 = WMMA_BF16(a1, b1, acc11);
    }
    const float inv_es = __expf(-scale[0]);
    const int col0 = n0 + l15, col1 = n0 + 16 + l15;
    const float k2v0 = k2[col0], k2v1 = k2[col1];
#pragma unroll
    for (int im = 0; im < 2; im++) {
#pragma unroll
        for (int i = 0; i < 8; i++) {
            int row = m0 + im * 16 + half * 8 + i;
            float q2v = q2[row];
            float Bb = 1.f - q2v;
#pragma unroll
            for (int in = 0; in < 2; in++) {
                float qk  = (im == 0) ? (in == 0 ? acc00[i] : acc01[i])
                                      : (in == 0 ? acc10[i] : acc11[i]);
                float k2v = (in == 0) ? k2v0 : k2v1;
                int col   = (in == 0) ? col0 : col1;
                float Aa = 1.f - 2.f * qk + k2v;
                float num2 = Aa * Aa * q2v - 2.f * Aa * Bb * qk + Bb * Bb * k2v;
                float den = fmaxf(fabsf(1.f - 2.f * qk + q2v * k2v), EPSF);
                float u = sqrtf(fmaxf(num2, 0.f)) / den;
                u = fminf(u, 1.f - BALL_EPSF);
                C[(size_t)row * S + col] = -2.f * atanh_f(u) * inv_es;
            }
        }
    }
}

// ---- E1: h = scalar_mul(sqrt(.5), mobius_add(tgt, mobius_add(matvec_epi(mx,x), b_in))) ----
__global__ __launch_bounds__(256)
void k_e1(const float* __restrict__ x, const float* __restrict__ tgt,
          const float* __restrict__ mx, const float* __restrict__ b_in,
          __bf16* __restrict__ hbf, float* __restrict__ q2out) {
    __shared__ float sm[256];
    const int row = blockIdx.x;
    const float* xr = x   + (size_t)row * 1024;
    const float* tr = tgt + (size_t)row * 1024;
    const float* mr = mx  + (size_t)row * 1024;
    float xs = 0.f, ms = 0.f;
    float tv[4], mv[4], bv[4];
#pragma unroll
    for (int j = 0; j < 4; j++) {
        int idx = threadIdx.x + j * 256;
        float xv = xr[idx]; tv[j] = tr[idx]; mv[j] = mr[idx]; bv[j] = b_in[idx];
        xs += xv * xv; ms += mv[j] * mv[j];
    }
    float xn  = fminf(fmaxf(sqrtf(blk_sum(xs, sm)), EPSF), 1.f - BALL_EPSF);
    float mxn = fmaxf(sqrtf(blk_sum(ms, sm)), EPSF);
    float fac = tanhf(mxn / xn * atanh_f(xn)) / mxn;
    float x2s = 0.f, y2s = 0.f, xys = 0.f;
#pragma unroll
    for (int j = 0; j < 4; j++) {
        mv[j] *= fac;
        x2s += mv[j] * mv[j]; y2s += bv[j] * bv[j]; xys += mv[j] * bv[j];
    }
    float x2 = blk_sum(x2s, sm), y2 = blk_sum(y2s, sm), xy = blk_sum(xys, sm);
    float c1 = 1.f + 2.f * xy + y2, c2 = 1.f - x2;
    float dn = fmaxf(1.f + 2.f * xy + x2 * y2, EPSF);
    float pv[4];
    x2s = 0.f; y2s = 0.f; xys = 0.f;
#pragma unroll
    for (int j = 0; j < 4; j++) {
        pv[j] = (c1 * mv[j] + c2 * bv[j]) / dn;
        x2s += tv[j] * tv[j]; y2s += pv[j] * pv[j]; xys += tv[j] * pv[j];
    }
    x2 = blk_sum(x2s, sm); y2 = blk_sum(y2s, sm); xy = blk_sum(xys, sm);
    c1 = 1.f + 2.f * xy + y2; c2 = 1.f - x2;
    dn = fmaxf(1.f + 2.f * xy + x2 * y2, EPSF);
    float gv[4]; float g2s = 0.f;
#pragma unroll
    for (int j = 0; j < 4; j++) { gv[j] = (c1 * tv[j] + c2 * pv[j]) / dn; g2s += gv[j] * gv[j]; }
    float gn = fminf(fmaxf(sqrtf(blk_sum(g2s, sm)), EPSF), 1.f - BALL_EPSF);
    float f2 = tanhf(0.70710678118654752f * atanh_f(gn)) / gn;
    float hs = 0.f;
#pragma unroll
    for (int j = 0; j < 4; j++) {
        int idx = threadIdx.x + j * 256;
        float h = f2 * gv[j];
        hbf[(size_t)row * 1024 + idx] = (__bf16)h;
        hs += h * h;
    }
    float h2 = blk_sum(hs, sm);
    if (threadIdx.x == 0) q2out[row] = h2;
}

// ---- softmax over S, writes fp32 attn (to d_out) + bf16 copy ----
__global__ __launch_bounds__(256)
void k_softmax(const float* __restrict__ Sc, const unsigned char* __restrict__ mask,
               float* __restrict__ attn, __bf16* __restrict__ attnbf) {
    __shared__ float sm[256];
    const int row = blockIdx.x;
    const int b = row >> 10;
    const float* sr = Sc + (size_t)row * 1024;
    const unsigned char* mrow = mask + (size_t)b * 1024;
    float v[4]; float mx = -INFINITY;
#pragma unroll
    for (int j = 0; j < 4; j++) {
        int idx = threadIdx.x + j * 256;
        float s = mrow[idx] ? -INFINITY : sr[idx];
        v[j] = s; mx = fmaxf(mx, s);
    }
    float rmax = blk_max(mx, sm);
    float ss = 0.f;
#pragma unroll
    for (int j = 0; j < 4; j++) { float e = __expf(v[j] - rmax); v[j] = e; ss += e; }
    float inv = 1.f / blk_sum(ss, sm);
#pragma unroll
    for (int j = 0; j < 4; j++) {
        int idx = threadIdx.x + j * 256;
        float a = v[j] * inv;
        attn  [(size_t)row * 1024 + idx] = a;
        attnbf[(size_t)row * 1024 + idx] = (__bf16)a;
    }
}

// ---- den[b,t] = sum_s attn * (lam-1) ----
__global__ __launch_bounds__(256)
void k_den(const float* __restrict__ attn, const float* __restrict__ lamm1,
           float* __restrict__ den) {
    __shared__ float sm[256];
    const int row = blockIdx.x; const int b = row >> 10;
    const float* ar = attn + (size_t)row * 1024;
    const float* lr = lamm1 + (size_t)b * 1024;
    float s = 0.f;
#pragma unroll
    for (int j = 0; j < 4; j++) { int idx = threadIdx.x + j * 256; s += ar[idx] * lr[idx]; }
    float d = blk_sum(s, sm);
    if (threadIdx.x == 0) { if (fabsf(d) < EPSF) d = EPSF; den[row] = d; }
}

// ---- E2: midpoint closed form + scalar_mul(sqrt(S - nmask)) -> o_bf, ||o|| ----
__global__ __launch_bounds__(256)
void k_e2(const float* __restrict__ nom, const float* __restrict__ den,
          const unsigned char* __restrict__ mask,
          __bf16* __restrict__ obf, float* __restrict__ onrm) {
    __shared__ float sm[256];
    const int row = blockIdx.x; const int b = row >> 10;
    const float d = den[row];
    const float* nr = nom + (size_t)row * 1024;
    const unsigned char* mrow = mask + (size_t)b * 1024;
    float tv[4]; float t2s = 0.f, msum = 0.f;
#pragma unroll
    for (int j = 0; j < 4; j++) {
        int idx = threadIdx.x + j * 256;
        tv[j] = nr[idx] / d;
        t2s += tv[j] * tv[j];
        msum += mrow[idx] ? 1.f : 0.f;
    }
    float tm2 = blk_sum(t2s, sm);
    float mcnt = blk_sum(msum, sm);
    float oc = 1.f / (1.f + sqrtf(fmaxf(1.f - tm2, EPSF)));
    float o2s = 0.f;
#pragma unroll
    for (int j = 0; j < 4; j++) { tv[j] *= oc; o2s += tv[j] * tv[j]; }
    float on = fminf(fmaxf(sqrtf(blk_sum(o2s, sm)), EPSF), 1.f - BALL_EPSF);
    float r = sqrtf(1024.f - mcnt);
    float f = tanhf(r * atanh_f(on)) / on;
    float n2s = 0.f;
#pragma unroll
    for (int j = 0; j < 4; j++) {
        int idx = threadIdx.x + j * 256;
        float o = f * tv[j];
        obf[(size_t)row * 1024 + idx] = (__bf16)o;
        n2s += o * o;
    }
    float on2 = blk_sum(n2s, sm);
    if (threadIdx.x == 0) onrm[row] = fminf(fmaxf(sqrtf(on2), EPSF), 1.f - BALL_EPSF);
}

// ---- E3: out = scalar_mul(sqrt(.5), mobius_add(x, mobius_add(matvec_epi(mo,o), b_out))) ----
__global__ __launch_bounds__(256)
void k_e3(const float* __restrict__ x, const float* __restrict__ mo,
          const float* __restrict__ b_out, const float* __restrict__ onrm,
          float* __restrict__ out) {
    __shared__ float sm[256];
    const int row = blockIdx.x;
    const float* xr = x  + (size_t)row * 1024;
    const float* mr = mo + (size_t)row * 1024;
    float xv[4], mv[4], bv[4]; float ms = 0.f;
#pragma unroll
    for (int j = 0; j < 4; j++) {
        int idx = threadIdx.x + j * 256;
        xv[j] = xr[idx]; mv[j] = mr[idx]; bv[j] = b_out[idx];
        ms += mv[j] * mv[j];
    }
    float mxn = fmaxf(sqrtf(blk_sum(ms, sm)), EPSF);
    float on = onrm[row];
    float fac = tanhf(mxn / on * atanh_f(on)) / mxn;
    float x2s = 0.f, y2s = 0.f, xys = 0.f;
#pragma unroll
    for (int j = 0; j < 4; j++) {
        mv[j] *= fac;
        x2s += mv[j] * mv[j]; y2s += bv[j] * bv[j]; xys += mv[j] * bv[j];
    }
    float x2 = blk_sum(x2s, sm), y2 = blk_sum(y2s, sm), xy = blk_sum(xys, sm);
    float c1 = 1.f + 2.f * xy + y2, c2 = 1.f - x2;
    float dn = fmaxf(1.f + 2.f * xy + x2 * y2, EPSF);
    float pv[4];
    x2s = 0.f; y2s = 0.f; xys = 0.f;
#pragma unroll
    for (int j = 0; j < 4; j++) {
        pv[j] = (c1 * mv[j] + c2 * bv[j]) / dn;
        x2s += xv[j] * xv[j]; y2s += pv[j] * pv[j]; xys += xv[j] * pv[j];
    }
    x2 = blk_sum(x2s, sm); y2 = blk_sum(y2s, sm); xy = blk_sum(xys, sm);
    c1 = 1.f + 2.f * xy + y2; c2 = 1.f - x2;
    dn = fmaxf(1.f + 2.f * xy + x2 * y2, EPSF);
    float gv[4]; float g2s = 0.f;
#pragma unroll
    for (int j = 0; j < 4; j++) { gv[j] = (c1 * xv[j] + c2 * pv[j]) / dn; g2s += gv[j] * gv[j]; }
    float gn = fminf(fmaxf(sqrtf(blk_sum(g2s, sm)), EPSF), 1.f - BALL_EPSF);
    float f2 = tanhf(0.70710678118654752f * atanh_f(gn)) / gn;
#pragma unroll
    for (int j = 0; j < 4; j++) {
        int idx = threadIdx.x + j * 256;
        out[(size_t)row * 1024 + idx] = f2 * gv[j];
    }
}

extern "C" void kernel_launch(void* const* d_in, const int* in_sizes, int n_in,
                              void* d_out, int out_size, void* d_ws, size_t ws_size,
                              hipStream_t stream) {
    const float* x     = (const float*)d_in[0];
    const float* tgt   = (const float*)d_in[1];
    const float* enck  = (const float*)d_in[2];
    const float* encv  = (const float*)d_in[3];
    const unsigned char* mask = (const unsigned char*)d_in[4];
    const float* W_in  = (const float*)d_in[5];
    const float* b_in  = (const float*)d_in[6];
    const float* W_out = (const float*)d_in[7];
    const float* b_out = (const float*)d_in[8];
    const float* scale = (const float*)d_in[9];

    constexpr int B = 8, T = 1024, S = 1024, E = 1024, C = 1024;
    const size_t MB = 1024 * 1024;
    char* ws = (char*)d_ws;
    float*  R0    = (float*)(ws);               // 32MB: mx, later nom
    float*  R1    = (float*)(ws + 32 * MB);     // 32MB: scores, later mo
    __bf16* hbf   = (__bf16*)(ws + 64 * MB);    // 16MB: h_bf, later o_bf
    __bf16* kbf   = (__bf16*)(ws + 80 * MB);    // 16MB
    __bf16* vlbf  = (__bf16*)(ws + 96 * MB);    // 16MB (row-major v*lam)
    __bf16* xbf   = (__bf16*)(ws + 112 * MB);   // 16MB: x_bf, later attn_bf
    __bf16* vlbT  = (__bf16*)(ws + 128 * MB);   // 16MB (transposed v*lam, [E,S] per batch)
    __bf16* wibf  = (__bf16*)(ws + 144 * MB);   // 2MB
    __bf16* wobf  = (__bf16*)(ws + 146 * MB);   // 2MB
    float*  q2    = (float*)(ws + 148 * MB);    // B*T
    float*  k2    = q2 + B * T;                 // B*S
    float*  lamm1 = k2 + B * S;                 // B*S
    float*  den   = lamm1 + B * S;              // B*T
    float*  onrm  = den + B * T;                // B*T

    float* outp = (float*)d_out;
    float* attn = outp + (size_t)B * T * E;

    const int thr = 256;
    k_cvt_bf16<<<4096, thr, 0, stream>>>(x, xbf, B * T * C);
    k_cvt_bf16<<<1024, thr, 0, stream>>>(W_in, wibf, E * C);
    k_cvt_bf16<<<1024, thr, 0, stream>>>(W_out, wobf, C * E);
    k_prep_kv<<<B * S, thr, 0, stream>>>(enck, encv, kbf, vlbf, k2, lamm1);
    dim3 gT(16, 16, B);
    k_transpose_bf<<<gT, thr, 0, stream>>>(vlbf, vlbT);

    dim3 gBig(E / 128, (B * T) / 64, 1);        // (8, 128)
    k_gemm_nt<<<gBig, thr, 0, stream>>>(xbf, wibf, R0, E, C, 0, 0, 0);      // mx
    k_e1<<<B * T, thr, 0, stream>>>(x, tgt, R0, b_in, hbf, q2);             // h_bf, q2

    dim3 gBat(S / 128, T / 64, B);              // (8, 16, 8)
    k_gemm_scores<<<gBat, thr, 0, stream>>>(hbf, kbf, q2, k2, scale, R1);
    k_softmax<<<B * T, thr, 0, stream>>>(R1, mask, attn, xbf);              // attn + attn_bf
    k_den<<<B * T, thr, 0, stream>>>(attn, lamm1, den);
    // nom[t,e] = sum_s attn[t,s] * vlamT[e,s]  -> NT with transposed operand
    dim3 gBat2(E / 128, T / 64, B);
    k_gemm_nt<<<gBat2, thr, 0, stream>>>(xbf, vlbT, R0, E, S,
                                         (size_t)T * S, (size_t)E * S, (size_t)T * E);
    k_e2<<<B * T, thr, 0, stream>>>(R0, den, mask, hbf, onrm);              // o_bf, ||o||
    k_gemm_nt<<<gBig, thr, 0, stream>>>(hbf, wobf, R1, C, E, 0, 0, 0);      // mo
    k_e3<<<B * T, thr, 0, stream>>>(x, R1, b_out, onrm, outp);

    (void)in_sizes; (void)n_in; (void)out_size; (void)ws_size;
}
